// GFSA_56650618635176
// MI455X (gfx1250) — compile-verified, matrix operand order checked by reading, not run on hardware
//
#include <hip/hip_runtime.h>

typedef _Float16 half_t;
typedef __attribute__((ext_vector_type(16))) _Float16 v16h;
typedef __attribute__((ext_vector_type(8)))  float    v8f;

#define WMMA_F16(a, b, c) \
  __builtin_amdgcn_wmma_f32_16x16x32_f16(false, (a), false, (b), (short)0, (c), false, false)

#define LDS_AS __attribute__((address_space(3)))

// 32-bit LDS byte offset of a generic pointer to a __shared__ object.
__device__ __forceinline__ unsigned gfsa_lds_off(const void* p) {
  return (unsigned)(unsigned long long)(LDS_AS const char*)p;
}

// CDNA5 async global->LDS copy, 16 bytes per lane. Tracked by ASYNCcnt.
__device__ __forceinline__ void gfsa_async_ld_lds_b128(void* lds_dst,
                                                       const void* gsrc) {
  asm volatile("global_load_async_to_lds_b128 %0, %1, off"
               :
               : "v"(gfsa_lds_off(lds_dst)), "v"(gsrc)
               : "memory");
}

__device__ __forceinline__ void gfsa_wait_async0() {
  asm volatile("s_wait_asynccnt 0x0" ::: "memory");
}

// ---------------------------------------------------------------------------
// Generic tiled GEMM: C[M,Nc] = A[M,Kd] * W[Kd,Nc] + bias, f16 WMMA, f32 acc.
// OUT_MODE 0: write f16 in head layout [B,H,N,d]  (for Q/K/V projections)
// OUT_MODE 1: write f32 row-major [M,Nc]          (final output projection)
// Block: 128 threads (4 waves), 64x64 tile, K step 32. Staging converts
// f32 -> f16, so it uses normal (converting) LDS stores.
// ---------------------------------------------------------------------------
template <typename AT, int OUT_MODE>
__global__ void gfsa_gemm(const AT* __restrict__ A, const float* __restrict__ W,
                          const float* __restrict__ bias,
                          half_t* __restrict__ outH, float* __restrict__ outF,
                          int M, int Kd, int Nc) {
  __shared__ half_t As[64 * 32];
  __shared__ half_t Bs[32 * 64];

  const int n0 = blockIdx.x * 64;
  const int m0 = blockIdx.y * 64;
  const int tid = threadIdx.x;
  const int lane = tid & 31;
  const int wave = tid >> 5;
  const int r  = lane & 15;      // row (A) / col (B) within fragment
  const int hi = lane >> 4;      // lane half selects K sub-range
  const int kb = hi * 8;

  const v8f vzero = {0.f, 0.f, 0.f, 0.f, 0.f, 0.f, 0.f, 0.f};
  v8f acc[4];
#pragma unroll
  for (int i = 0; i < 4; ++i) acc[i] = vzero;

  for (int k0 = 0; k0 < Kd; k0 += 32) {
    // Stage A tile 64x32 (convert to f16)
    for (int e = tid; e < 64 * 32; e += 128) {
      const int rr = e >> 5, cc = e & 31;
      As[e] = (half_t)A[(size_t)(m0 + rr) * Kd + k0 + cc];
    }
    // Stage W tile 32x64 (convert to f16)
    for (int e = tid; e < 32 * 64; e += 128) {
      const int rr = e >> 6, cc = e & 63;
      Bs[e] = (half_t)W[(size_t)(k0 + rr) * Nc + n0 + cc];
    }
    __syncthreads();

    // A fragment: 16x32, one row per lane; K = kb..kb+7 and kb+16..kb+23
    v16h af;
    const int arow = wave * 16 + r;
#pragma unroll
    for (int i = 0; i < 8; ++i) {
      af[i]     = As[arow * 32 + kb + i];
      af[i + 8] = As[arow * 32 + kb + 16 + i];
    }
#pragma unroll
    for (int jn = 0; jn < 4; ++jn) {
      v16h bf;  // B fragment: 32x16, one column per lane
      const int bcol = jn * 16 + r;
#pragma unroll
      for (int i = 0; i < 8; ++i) {
        bf[i]     = Bs[(kb + i) * 64 + bcol];
        bf[i + 8] = Bs[(kb + 16 + i) * 64 + bcol];
      }
      acc[jn] = WMMA_F16(af, bf, acc[jn]);
    }
    __syncthreads();
  }

  // Epilogue: C element (VGPR v) -> row = v + 8*hi, col = lane&15
#pragma unroll
  for (int jn = 0; jn < 4; ++jn) {
    const int nc = n0 + jn * 16 + r;
    const float bval = bias[nc];
#pragma unroll
    for (int v = 0; v < 8; ++v) {
      const int m = m0 + wave * 16 + v + 8 * hi;
      const float val = acc[jn][v] + bval;
      if (OUT_MODE == 0) {
        const int b = m >> 10, n = m & 1023;       // m = b*N + n (N=1024)
        const int h = nc >> 6, dd = nc & 63;       // nc = h*d + dd (d=64)
        outH[((size_t)(b * 16 + h) * 1024 + n) * 64 + dd] = (half_t)val;
      } else {
        outF[(size_t)m * Nc + nc] = val;
      }
    }
  }
}

// ---------------------------------------------------------------------------
// Attention scores + softmax. One wave handles 16 query rows of one (b,h).
// Scores staged in LDS (col-major [1024][16], 64 KB), softmax over 1024,
// f16 att written to workspace.
// ---------------------------------------------------------------------------
__global__ void gfsa_attn(const half_t* __restrict__ Qh,
                          const half_t* __restrict__ Kh,
                          half_t* __restrict__ att) {
  __shared__ float sc[1024 * 16];  // sc[col * 16 + row]

  const int rb = blockIdx.x;   // query row block (0..63)
  const int bh = blockIdx.y;   // fused batch*head (0..63)
  const int lane = threadIdx.x;
  const int r  = lane & 15;
  const int hi = lane >> 4;
  const int kb = hi * 8;

  const size_t qkbase  = (size_t)bh * 1024 * 64;
  const size_t attbase = (size_t)bh * 1024 * 1024;

  // Q fragments for rows rb*16..+15: 16x64 split as two 16x32 A-fragments
  v16h a0, a1;
  {
    const size_t ro = qkbase + (size_t)(rb * 16 + r) * 64;
#pragma unroll
    for (int i = 0; i < 8; ++i) {
      a0[i]     = Qh[ro + kb + i];
      a0[i + 8] = Qh[ro + kb + 16 + i];
      a1[i]     = Qh[ro + kb + 32 + i];
      a1[i + 8] = Qh[ro + kb + 48 + i];
    }
  }

  const v8f vzero = {0.f, 0.f, 0.f, 0.f, 0.f, 0.f, 0.f, 0.f};
  for (int jt = 0; jt < 64; ++jt) {
    // B = K^T tile: B[kk, n] = Kh[jt*16 + n, kk] (row-major Kh -> contiguous)
    v16h b0, b1;
    const size_t co = qkbase + (size_t)(jt * 16 + r) * 64;
#pragma unroll
    for (int i = 0; i < 8; ++i) {
      b0[i]     = Kh[co + kb + i];
      b0[i + 8] = Kh[co + kb + 16 + i];
      b1[i]     = Kh[co + kb + 32 + i];
      b1[i + 8] = Kh[co + kb + 48 + i];
    }
    v8f acc = vzero;
    acc = WMMA_F16(a0, b0, acc);
    acc = WMMA_F16(a1, b1, acc);
#pragma unroll
    for (int v = 0; v < 8; ++v)
      sc[(jt * 16 + r) * 16 + (v + 8 * hi)] = acc[v] * 0.125f;  // 1/sqrt(64)
  }
  __syncthreads();

  // Softmax: row = lane&15; the two lane-halves split the 1024 columns.
  float mx = -1e30f;
  for (int c = 0; c < 512; ++c)
    mx = fmaxf(mx, sc[(hi * 512 + c) * 16 + r]);
  mx = fmaxf(mx, __shfl_xor(mx, 16, 32));

  float sum = 0.f;
  for (int c = 0; c < 512; ++c)
    sum += __expf(sc[(hi * 512 + c) * 16 + r] - mx);
  sum += __shfl_xor(sum, 16, 32);
  const float inv = 1.0f / sum;

  for (int c = 0; c < 512; ++c) {
    const int col = hi * 512 + c;
    const float e = __expf(sc[col * 16 + r] - mx) * inv;
    att[attbase + (size_t)(rb * 16 + r) * 1024 + col] = (half_t)e;
  }
}

// ---------------------------------------------------------------------------
// T[N,64] = att[N,N] * Bm[N,64] per (b,h).
// B tile (32x64 f16, 4 KB) is staged into LDS once per K-step with CDNA5
// async global->LDS b128 copies, shared by all 4 waves of the block.
// combine==0: write T1 (f16, head layout).
// combine==1: fused epilogue  OH = w0*V + (w1-wK)*T1 + 2*wK*T2  in [B,N,D].
// Block: 128 threads (4 waves); each wave does 16 rows x 64 cols.
// ---------------------------------------------------------------------------
__global__ void gfsa_attmm(const half_t* __restrict__ att,
                           const half_t* __restrict__ Bm,
                           const half_t* __restrict__ Vh,
                           const half_t* __restrict__ T1,
                           const float* __restrict__ w0,
                           const float* __restrict__ w1,
                           const float* __restrict__ wK,
                           half_t* __restrict__ outT, half_t* __restrict__ OH,
                           int combine) {
  __shared__ __align__(16) half_t Bs[32 * 64];

  const int rb = blockIdx.x;   // 0..15, 64 rows per block
  const int bh = blockIdx.y;   // 0..63
  const int tid = threadIdx.x;
  const int lane = tid & 31;
  const int wave = tid >> 5;
  const int r  = lane & 15;
  const int hi = lane >> 4;
  const int kb = hi * 8;

  const size_t attbase = (size_t)bh * 1024 * 1024;
  const size_t ndbase  = (size_t)bh * 1024 * 64;
  const int rowbase = rb * 64 + wave * 16;

  const v8f vzero = {0.f, 0.f, 0.f, 0.f, 0.f, 0.f, 0.f, 0.f};
  v8f acc[4];
#pragma unroll
  for (int i = 0; i < 4; ++i) acc[i] = vzero;

  for (int k0 = 0; k0 < 1024; k0 += 32) {
    // Async-stage B tile: 32x64 f16 = 256 x 16B chunks, 2 chunks per thread.
#pragma unroll
    for (int j = 0; j < 2; ++j) {
      const int c = tid + j * 128;            // chunk id (8 halves each)
      const int brow = c >> 3;                // 8 chunks per 64-half row
      const int bcol8 = (c & 7) * 8;
      gfsa_async_ld_lds_b128(&Bs[c * 8],
                             &Bm[ndbase + (size_t)(k0 + brow) * 64 + bcol8]);
    }
    gfsa_wait_async0();   // own async copies landed in LDS
    __syncthreads();      // all waves' copies landed

    v16h af;
    const size_t ao = attbase + (size_t)(rowbase + r) * 1024 + k0;
    if (k0 + 32 < 1024)
      __builtin_prefetch((const void*)(att + ao + 32), 0, 0);
#pragma unroll
    for (int i = 0; i < 8; ++i) {
      af[i]     = att[ao + kb + i];
      af[i + 8] = att[ao + kb + 16 + i];
    }
#pragma unroll
    for (int jn = 0; jn < 4; ++jn) {
      v16h bf;
      const int bcol = jn * 16 + r;
#pragma unroll
      for (int i = 0; i < 8; ++i) {
        bf[i]     = Bs[(kb + i) * 64 + bcol];
        bf[i + 8] = Bs[(kb + 16 + i) * 64 + bcol];
      }
      acc[jn] = WMMA_F16(af, bf, acc[jn]);
    }
    __syncthreads();      // before next K-step overwrites Bs
  }

  const int h = bh & 15, b = bh >> 4;
  float c0 = 0.f, c1 = 0.f, c2 = 0.f;
  if (combine) {
    const float W0 = w0[h], W1 = w1[h], WKv = wK[h];
    c0 = W0; c1 = W1 - WKv; c2 = 2.f * WKv;  // K_POLY = 3
  }
#pragma unroll
  for (int jn = 0; jn < 4; ++jn) {
#pragma unroll
    for (int v = 0; v < 8; ++v) {
      const int row = rowbase + v + 8 * hi;
      const int col = jn * 16 + r;
      if (!combine) {
        outT[ndbase + (size_t)row * 64 + col] = (half_t)acc[jn][v];
      } else {
        const size_t idx = ndbase + (size_t)row * 64 + col;
        const float vv = c0 * (float)Vh[idx] + c1 * (float)T1[idx] +
                         c2 * acc[jn][v];
        OH[((size_t)b * 1024 + row) * 1024 + h * 64 + col] = (half_t)vv;
      }
    }
  }
}

// ---------------------------------------------------------------------------
// Host orchestration. Workspace layout (bytes):
//   [0,8M)   Qh f16 [64,1024,64]     [8M,16M)  Kh      [16M,24M) Vh
//   [24M,32M) T1 f16                 [32M,40M) OH f16 [4096,1024]
//   [40M,168M) att f16 [64,1024,1024]
// ---------------------------------------------------------------------------
extern "C" void kernel_launch(void* const* d_in, const int* in_sizes, int n_in,
                              void* d_out, int out_size, void* d_ws,
                              size_t ws_size, hipStream_t stream) {
  const float* x  = (const float*)d_in[0];
  const float* Wq = (const float*)d_in[1];
  const float* bq = (const float*)d_in[2];
  const float* Wk = (const float*)d_in[3];
  const float* bk = (const float*)d_in[4];
  const float* Wv = (const float*)d_in[5];
  const float* bv = (const float*)d_in[6];
  const float* Wo = (const float*)d_in[7];
  const float* bo = (const float*)d_in[8];
  const float* w0 = (const float*)d_in[9];
  const float* w1 = (const float*)d_in[10];
  const float* wK = (const float*)d_in[11];
  float* out = (float*)d_out;

  char* ws = (char*)d_ws;
  const size_t MB = 1u << 20;
  half_t* Qh  = (half_t*)(ws + 0 * MB);
  half_t* Kh  = (half_t*)(ws + 8 * MB);
  half_t* Vh  = (half_t*)(ws + 16 * MB);
  half_t* T1  = (half_t*)(ws + 24 * MB);
  half_t* OH  = (half_t*)(ws + 32 * MB);
  half_t* att = (half_t*)(ws + 40 * MB);

  const dim3 gG(16, 64), bG(128);
  // Q/K/V projections (f32 -> f16 head layout)
  gfsa_gemm<float, 0><<<gG, bG, 0, stream>>>(x, Wq, bq, Qh, nullptr, 4096, 1024, 1024);
  gfsa_gemm<float, 0><<<gG, bG, 0, stream>>>(x, Wk, bk, Kh, nullptr, 4096, 1024, 1024);
  gfsa_gemm<float, 0><<<gG, bG, 0, stream>>>(x, Wv, bv, Vh, nullptr, 4096, 1024, 1024);
  // att = softmax(q k^T / 8)
  gfsa_attn<<<dim3(64, 64), 32, 0, stream>>>(Qh, Kh, att);
  // T1 = att @ V ; OH = w0*V + (w1-wK)*T1 + 2wK*(att @ T1)
  gfsa_attmm<<<dim3(16, 64), 128, 0, stream>>>(att, Vh, Vh, T1, w0, w1, wK, T1, nullptr, 0);
  gfsa_attmm<<<dim3(16, 64), 128, 0, stream>>>(att, T1, Vh, T1, w0, w1, wK, nullptr, OH, 1);
  // out = OH @ Wo + bo (f32)
  gfsa_gemm<half_t, 1><<<gG, bG, 0, stream>>>(OH, Wo, bo, nullptr, out, 4096, 1024, 1024);
}
